// AttentionHeadRankFour_41197326303952
// MI455X (gfx1250) — matrix operand order repaired and checked by reading, not compiled, not run on hardware
//
#include <hip/hip_runtime.h>
#include <stdint.h>

// Problem constants: B=8, X=8 -> H=64 heads; S=1024; D_IN=512; D_OUT=64
#define H_HEADS 64
#define SEQ     1024
#define DIN     512
#define DM      64

typedef __attribute__((ext_vector_type(16))) __bf16 v16bf;
typedef __attribute__((ext_vector_type(8)))  float  v8f;

union ABTile { v16bf v; uint32_t u[8]; uint4 q[2]; };

// Native bf16 converts (RNE in HW) -> v_cvt_pk_bf16_f32 class codegen instead
// of a 6-op manual rounding sequence per element.
__device__ __forceinline__ uint32_t cvt_pk_bf16(float a, float b) {
    union { __bf16 h[2]; uint32_t u; } pk;
    pk.h[0] = (__bf16)a;
    pk.h[1] = (__bf16)b;
    return pk.u;
}
__device__ __forceinline__ unsigned short f2bf(float a) {
    union { __bf16 h; unsigned short s; } c;
    c.h = (__bf16)a;
    return c.s;
}
// Same-wave LDS RAW/WAR ordering (per-wave private slabs; no workgroup barrier
// possible because waves have divergent causal loop trip counts).
__device__ __forceinline__ void lds_fence() {
    asm volatile("s_wait_dscnt 0" ::: "memory");
}

// -------------------------------------------------------------------------
// Kernel 0: pack the three 512x64 f32 weights into bf16, laid out exactly as
// WMMA B-operand tiles: linear index = (((p*4 + t)*16 + s)*32 + lane)*8 + d
// where element dword d of lane L holds W[k][n], W[k+1][n] with
// k = s*32 + (L>>4)*16 + 2d, n = t*16 + (L&15).
// -------------------------------------------------------------------------
__global__ __launch_bounds__(256) void pack_weights_kernel(
    const float* __restrict__ kw, const float* __restrict__ vw,
    const float* __restrict__ qw, uint32_t* __restrict__ wpk)
{
    int idx = blockIdx.x * 256 + threadIdx.x;
    if (idx >= 3 * 4 * 16 * 32 * 8) return;
    int d    = idx & 7;
    int lane = (idx >> 3) & 31;
    int s    = (idx >> 8) & 15;
    int t    = (idx >> 12) & 3;
    int p    = idx >> 14;
    const float* w = (p == 0) ? kw : (p == 1) ? vw : qw;
    int kk  = s * 32 + (lane >> 4) * 16 + 2 * d;
    int col = t * 16 + (lane & 15);
    wpk[idx] = cvt_pk_bf16(w[kk * DM + col], w[(kk + 1) * DM + col]);
}

// -------------------------------------------------------------------------
// Kernel 1: fused QKV projection. 512 blocks x 256 threads (8 waves).
// Each wave: 16 rows x 64 cols per projection, K=512 in 16 steps of 32.
// Outputs: K,Q row-major bf16 [row][64]; V transposed bf16 [head][64][1024].
// -------------------------------------------------------------------------
__global__ __launch_bounds__(256) void proj_kernel(
    const float* __restrict__ inK, const float* __restrict__ inV,
    const float* __restrict__ inQ, const uint32_t* __restrict__ wpk,
    unsigned short* __restrict__ qb, unsigned short* __restrict__ kb_o,
    unsigned short* __restrict__ vt)
{
    const int w    = threadIdx.x >> 5;
    const int lane = threadIdx.x & 31;
    const int half = lane >> 4;
    const int m    = lane & 15;
    const int nn   = lane & 15;
    const int row  = blockIdx.x * 128 + w * 16 + m;   // 0..65535

    for (int p = 0; p < 3; ++p) {
        const float* in = (p == 0) ? inK : (p == 1) ? inV : inQ;
        v8f acc0 = {}, acc1 = {}, acc2 = {}, acc3 = {};
        for (int s = 0; s < 16; ++s) {
            const int kbase = s * 32;
            // A tile: 16x32 bf16, contiguous layout: lane half h, dword d holds
            // K = kbase + h*16 + d*2  (pair, contiguous in mem)
            ABTile a;
            const float* arow = in + (size_t)row * DIN + kbase + half * 16;
#pragma unroll
            for (int d = 0; d < 8; ++d) {
                int kd = 2 * d;
                float2 f = *(const float2*)(arow + kd);
                a.u[d] = cvt_pk_bf16(f.x, f.y);
            }
            // Four pre-packed B tiles (L2-resident), 2x b128 loads each.
#pragma unroll
            for (int t = 0; t < 4; ++t) {
                ABTile b;
                const uint4* bp =
                    (const uint4*)(wpk + ((((p * 4 + t) * 16 + s) * 32 + lane) * 8));
                b.q[0] = bp[0];
                b.q[1] = bp[1];
                v8f& acc = (t == 0) ? acc0 : (t == 1) ? acc1 : (t == 2) ? acc2 : acc3;
                acc = __builtin_amdgcn_wmma_f32_16x16x32_bf16(
                    false, a.v, false, b.v, (short)0, acc, false, false);
            }
        }
        // Store C tiles (lane holds rows i+8*half, col t*16+nn) as bf16.
#pragma unroll
        for (int t = 0; t < 4; ++t) {
            const v8f acc = (t == 0) ? acc0 : (t == 1) ? acc1 : (t == 2) ? acc2 : acc3;
#pragma unroll
            for (int i = 0; i < 8; ++i) {
                int rg  = blockIdx.x * 128 + w * 16 + i + 8 * half;
                int col = t * 16 + nn;
                unsigned short hv = f2bf(acc[i]);
                if (p == 0) {
                    kb_o[(size_t)rg * DM + col] = hv;
                } else if (p == 2) {
                    qb[(size_t)rg * DM + col] = hv;
                } else {                       // V stored transposed per head
                    int head = rg >> 10, sr = rg & 1023;
                    vt[((size_t)head * DM + col) * SEQ + sr] = hv;
                }
            }
        }
    }
}

// -------------------------------------------------------------------------
// Kernel 2: causal flash attention. 512 blocks (head x q-block of 128),
// 256 threads = 8 waves; each wave owns 16 q rows, streams key blocks of 32.
// Softmax uses the full wave: lane L and L+16 split row (L&15) into two
// 16-column halves, combined with __shfl_xor(…, 16).
// -------------------------------------------------------------------------
__global__ __launch_bounds__(256) void attn_kernel(
    const unsigned short* __restrict__ qb, const unsigned short* __restrict__ kb,
    const unsigned short* __restrict__ vt, float* __restrict__ out)
{
    __shared__ float          sc_s[8][16 * 32];   // score tile per wave (f32)
    __shared__ unsigned short p_s[8][16 * 32];    // P tile per wave (bf16, row-major)
    __shared__ float          rs_s[8][16];        // per-row rescale broadcast

    const int w    = threadIdx.x >> 5;
    const int lane = threadIdx.x & 31;
    const int half = lane >> 4;
    const int m    = lane & 15;
    const int nn   = lane & 15;
    const int head = blockIdx.x >> 3;
    const int qi0  = (blockIdx.x & 7) * 128 + w * 16;

    // Q A-tiles (16 rows x 64 dims), loaded once: qa0 = d[0..31], qa1 = d[32..63]
    ABTile qa0, qa1;
    {
        const unsigned short* qrow =
            qb + ((size_t)head * SEQ + qi0 + m) * DM + half * 16;
#pragma unroll
        for (int d = 0; d < 8; ++d) {
            int kd = 2 * d;
            qa0.u[d] = *(const uint32_t*)(qrow + kd);
            qa1.u[d] = *(const uint32_t*)(qrow + 32 + kd);
        }
    }

    v8f o0 = {}, o1 = {}, o2 = {}, o3 = {};
    // Online-softmax row state, replicated across the lane pair (L, L+16).
    float m_i = -3.0e38f, l_i = 0.0f;

    const int kb_end = ((qi0 + 15) >> 5) + 1;   // causal: skip fully-masked blocks
    for (int kblk = 0; kblk < kb_end; ++kblk) {
        // ---- S tile = Q @ K^T for 32 keys (two 16x16 n-tiles) ----
#pragma unroll
        for (int nt = 0; nt < 2; ++nt) {
            const unsigned short* krow =
                kb + ((size_t)head * SEQ + kblk * 32 + nt * 16 + nn) * DM + half * 16;
            ABTile b0, b1;
#pragma unroll
            for (int d = 0; d < 8; ++d) {
                b0.u[d] = *(const uint32_t*)(krow + 2 * d);        // dims 0..31
                b1.u[d] = *(const uint32_t*)(krow + 32 + 2 * d);   // dims 32..63
            }
            v8f sacc = {};
            sacc = __builtin_amdgcn_wmma_f32_16x16x32_bf16(
                false, qa0.v, false, b0.v, (short)0, sacc, false, false);
            sacc = __builtin_amdgcn_wmma_f32_16x16x32_bf16(
                false, qa1.v, false, b1.v, (short)0, sacc, false, false);
#pragma unroll
            for (int i = 0; i < 8; ++i)
                sc_s[w][(i + 8 * half) * 32 + nt * 16 + nn] = sacc[i];
        }
        lds_fence();

        // ---- online softmax: full wave, 2 lanes per row, 16 cols each ----
        {
            const int row = m;
            const int qg  = qi0 + row;
            float pv[16];
            float mt = -3.0e38f;
#pragma unroll
            for (int c = 0; c < 16; ++c) {
                int col = half * 16 + c;
                float v = sc_s[w][row * 32 + col] * 0.03125f;  // 1/sqrt(S=1024)
                if (kblk * 32 + col > qg) v = -3.0e38f;        // causal mask
                pv[c] = v;
                mt = fmaxf(mt, v);
            }
            mt = fmaxf(mt, __shfl_xor(mt, 16, 32));            // pair-combine max
            float mn = fmaxf(m_i, mt);
            float sf = __expf(m_i - mn);
            float sum = 0.0f;
#pragma unroll
            for (int c = 0; c < 16; ++c) {
                float e = __expf(pv[c] - mn);
                sum += e;
                pv[c] = e;
            }
            sum += __shfl_xor(sum, 16, 32);                    // pair-combine sum
            l_i = l_i * sf + sum;
            m_i = mn;
            if (half == 0) rs_s[w][row] = sf;
            uint32_t* pd = (uint32_t*)p_s[w];
#pragma unroll
            for (int c2 = 0; c2 < 8; ++c2)
                pd[row * 16 + half * 8 + c2] = cvt_pk_bf16(pv[2 * c2], pv[2 * c2 + 1]);
        }
        lds_fence();

        // ---- rescale running O by exp(m_old - m_new) per row ----
#pragma unroll
        for (int i = 0; i < 8; ++i) {
            float r = rs_s[w][i + 8 * half];
            o0[i] *= r; o1[i] *= r; o2[i] *= r; o3[i] *= r;
        }

        // ---- P as WMMA A-operand from LDS (contraction = 32 keys) ----
        ABTile pa;
        {
            const unsigned short* pr = p_s[w] + m * 32 + half * 16;
#pragma unroll
            for (int d = 0; d < 8; ++d) {
                int kd = 2 * d;
                pa.u[d] = *(const uint32_t*)(pr + kd);
            }
        }

        // ---- O += P @ V  (V transposed: contiguous key pairs per lane) ----
#pragma unroll
        for (int t = 0; t < 4; ++t) {
            const unsigned short* vrow =
                vt + ((size_t)head * DM + t * 16 + nn) * SEQ + kblk * 32 + half * 16;
            ABTile b;
#pragma unroll
            for (int d = 0; d < 8; ++d)
                b.u[d] = *(const uint32_t*)(vrow + 2 * d);
            v8f& o = (t == 0) ? o0 : (t == 1) ? o1 : (t == 2) ? o2 : o3;
            o = __builtin_amdgcn_wmma_f32_16x16x32_bf16(
                false, pa.v, false, b.v, (short)0, o, false, false);
        }
        lds_fence();   // slabs are reused next iteration
    }

    // ---- normalize by row sums and store fp32 output ----
    if (lane < 16) rs_s[w][lane] = 1.0f / l_i;   // lane == row for lanes 0..15
    lds_fence();
#pragma unroll
    for (int i = 0; i < 8; ++i) {
        float r = rs_s[w][i + 8 * half];
        int rg = qi0 + i + 8 * half;
        float* orow = out + ((size_t)head * SEQ + rg) * DM + nn;
        orow[0]  = o0[i] * r;
        orow[16] = o1[i] * r;
        orow[32] = o2[i] * r;
        orow[48] = o3[i] * r;
    }
}

// -------------------------------------------------------------------------
extern "C" void kernel_launch(void* const* d_in, const int* in_sizes, int n_in,
                              void* d_out, int out_size, void* d_ws, size_t ws_size,
                              hipStream_t stream)
{
    const float* inK = (const float*)d_in[0];
    const float* inV = (const float*)d_in[1];
    const float* inQ = (const float*)d_in[2];
    const float* kw  = (const float*)d_in[3];
    const float* vw  = (const float*)d_in[4];
    const float* qw  = (const float*)d_in[5];

    // Workspace layout: Q bf16 (8MB) | K bf16 (8MB) | V^T bf16 (8MB) | packed W (192KB)
    char* ws = (char*)d_ws;
    unsigned short* qb  = (unsigned short*)(ws);
    unsigned short* kbb = (unsigned short*)(ws + (8u << 20));
    unsigned short* vt  = (unsigned short*)(ws + (16u << 20));
    uint32_t*       wpk = (uint32_t*)(ws + (24u << 20));

    pack_weights_kernel<<<192, 256, 0, stream>>>(kw, vw, qw, wpk);
    proj_kernel<<<512, 256, 0, stream>>>(inK, inV, inQ, wpk, qb, kbb, vt);
    attn_kernel<<<512, 256, 0, stream>>>(qb, kbb, vt, (float*)d_out);
}
